// LinearRNN_Block_48576080118641
// MI455X (gfx1250) — matrix operand order repaired and checked by reading, not compiled, not run on hardware
//
#include <hip/hip_runtime.h>

// ---------------------------------------------------------------------------
// Problem constants (from reference): B=8, S=2048, D_IN=512, D_H=1024
// ---------------------------------------------------------------------------
#define S_LEN 2048
#define B_SZ  8
#define DIN   512
#define DH    1024
#define ROWS  (B_SZ * S_LEN)   // 16384
#define NT_H  (DH / 16)        // 64 N-tiles
#define SCAN_WGS 16            // persistent workgroups in the scan

typedef __bf16 v8bf  __attribute__((ext_vector_type(8)));
typedef __bf16 v16bf __attribute__((ext_vector_type(16)));
typedef float  v8f   __attribute__((ext_vector_type(8)));

__device__ __forceinline__ unsigned short f2bf_rne(float f) {
    unsigned u = __float_as_uint(f);
    unsigned r = u + 0x7FFFu + ((u >> 16) & 1u);   // round-to-nearest-even
    return (unsigned short)(r >> 16);
}

// ---------------------------------------------------------------------------
// fp32 -> bf16 bulk convert (for x)
// ---------------------------------------------------------------------------
__global__ void k_f2bf(const float* __restrict__ in,
                       unsigned short* __restrict__ out, int n) {
    int stride = gridDim.x * blockDim.x;
    for (int i = blockIdx.x * blockDim.x + threadIdx.x; i < n; i += stride)
        out[i] = f2bf_rne(in[i]);
}

// ---------------------------------------------------------------------------
// Swizzle a weight matrix W[DH][K] (used as B-matrix, N=DH) into the CDNA5
// per-lane WMMA layout: Bsw[kc][nt][lane][16 halves], where lane<16 holds
// K = kc*32 + {0..7,16..23} of column nt*16+lane, lane>=16 holds K = kc*32 + {8..15,24..31}.
// ---------------------------------------------------------------------------
__global__ void k_swizzleB(const float* __restrict__ W,
                           unsigned short* __restrict__ Bsw, int K) {
    int n = K * DH;
    int stride = gridDim.x * blockDim.x;
    for (int idx = blockIdx.x * blockDim.x + threadIdx.x; idx < n; idx += stride) {
        int j    = idx & 15;
        int lane = (idx >> 4) & 31;
        int rest = idx >> 9;
        int nt   = rest & (NT_H - 1);
        int kc   = rest >> 6;
        int h    = nt * 16 + (lane & 15);
        int k    = kc * 32 + ((lane & 16) ? 8 : 0) + ((j & 8) ? 16 + (j - 8) : j);
        Bsw[idx] = f2bf_rne(W[(size_t)h * K + k]);
    }
}

// ---------------------------------------------------------------------------
// Zero the per-step barrier counters and the h exchange buffers (ws is
// poisoned with 0xAA by the harness; must re-zero every call).
// ---------------------------------------------------------------------------
__global__ void k_init(unsigned* __restrict__ cnt,
                       unsigned short* __restrict__ gbuf) {
    int stride = gridDim.x * blockDim.x;
    int i0 = blockIdx.x * blockDim.x + threadIdx.x;
    for (int i = i0; i < S_LEN; i += stride) cnt[i] = 0u;
    for (int i = i0; i < 2 * B_SZ * DH; i += stride) gbuf[i] = 0;
}

// ---------------------------------------------------------------------------
// WMMA GEMM + bias:  out[ROWS][DH] = Abf[ROWS][K] * W^T + bias
// Abf is bf16 row-major, Bsw is the pre-swizzled bf16 weight.
// One 16x16 output tile per wave, 4 waves (128 threads) per block.
// Operands stream from L2 (x + W are L2-resident on MI455X's 192MB L2).
// ---------------------------------------------------------------------------
__global__ void k_gemm_bias(const unsigned short* __restrict__ Abf,
                            const unsigned short* __restrict__ Bsw,
                            const float* __restrict__ bias,
                            float* __restrict__ out, int K) {
    const int lane = threadIdx.x & 31;
    const int wid  = threadIdx.x >> 5;
    const int tile = blockIdx.x * 4 + wid;
    const int mt = tile >> 6;      // / NT_H
    const int nt = tile & (NT_H - 1);
    const int KC = K >> 5;

    const int row = mt * 16 + (lane & 15);          // A: lane = M for all lanes
    const unsigned short* arow = Abf + (size_t)row * K;
    const int koff0 = (lane & 16) ? 8 : 0;          // K-halves split per lane group
    const int koff1 = koff0 + 16;

    v8f acc = {};
    for (int kc = 0; kc < KC; ++kc) {
        union { v16bf v; v8bf h[2]; } a;
        a.h[0] = *(const v8bf*)(arow + kc * 32 + koff0);
        a.h[1] = *(const v8bf*)(arow + kc * 32 + koff1);
        v16bf b = *(const v16bf*)(Bsw + ((size_t)(kc * NT_H + nt) * 32 + lane) * 16);
        acc = __builtin_amdgcn_wmma_f32_16x16x32_bf16(false, a.v, false, b,
                                                      (short)0, acc, false, false);
    }
    const int col = nt * 16 + (lane & 15);
    const int r0  = mt * 16 + ((lane & 16) ? 8 : 0);
    const float bs = bias[col];
#pragma unroll
    for (int i = 0; i < 8; ++i)
        out[(size_t)(r0 + i) * DH + col] = acc[i] + bs;
}

// ---------------------------------------------------------------------------
// Persistent scan kernel: 16 WGs, each owns a 64-column slice of W_A^T held
// in LDS (128KB bf16, only possible with CDNA5's 320KB WGP LDS).
// Per step: h_t[:, slice] = bx_t[:, slice] + h_{t-1} @ WA_slice  (WMMA, M=8
// padded to 16), slice exchange via double-buffered global bf16 buffer,
// cross-WG sync via per-step atomic counter.  y = h + residual written
// directly into d_out (residual pre-stored there by the second GEMM).
// ---------------------------------------------------------------------------
__global__ void k_scan(const unsigned short* __restrict__ WAsw,  // [32][64][32][16]
                       const float* __restrict__ bx,             // [ROWS][DH]
                       float* __restrict__ y,                    // d_out (residual in)
                       unsigned short* __restrict__ gbuf,        // [2][B_SZ][DH]
                       unsigned* __restrict__ cnt) {             // [S_LEN]
    extern __shared__ unsigned short smem[];
    unsigned short* bs = smem;                    // W_A slice: 32*4*32*16 = 65536 halves (128KB)
    unsigned short* hb = smem + 32 * 4 * 32 * 16; // h_{t-1}:   16*1024     = 16384 halves (32KB)

    const int tid  = threadIdx.x;     // 128 threads = 4 waves
    const int lane = tid & 31;
    const int wid  = tid >> 5;        // 0..3  -> local n-tile
    const int wg   = blockIdx.x;      // 0..15 -> 64-column slice

    // Stage this WG's W_A slice into LDS (4KB contiguous per k-chunk).
    for (int kc = 0; kc < 32; ++kc) {
        const uint4* src = (const uint4*)(WAsw + ((size_t)(kc * NT_H + wg * 4) * 32) * 16);
        uint4* dst = (uint4*)(bs + kc * 2048);
        for (int i = tid; i < 256; i += 128) dst[i] = src[i];
    }
    { // h_{-1} = 0 (rows 8..15 stay zero forever -> M padding contributes 0)
        uint4 z = {0u, 0u, 0u, 0u};
        uint4* hz = (uint4*)hb;
        for (int i = tid; i < 2048; i += 128) hz[i] = z;
    }
    __syncthreads();

    const int koff0 = (lane & 16) ? 8 : 0;
    const int koff1 = koff0 + 16;
    const unsigned short* hrow = hb + (size_t)(lane & 15) * DH;
    const int col = wg * 64 + wid * 16 + (lane & 15);

    for (int t = 0; t < S_LEN; ++t) {
        v8f acc = {};
#pragma unroll 4
        for (int kc = 0; kc < 32; ++kc) {
            union { v16bf v; v8bf h[2]; } a;
            a.h[0] = *(const v8bf*)(hrow + kc * 32 + koff0);
            a.h[1] = *(const v8bf*)(hrow + kc * 32 + koff1);
            v16bf b = *(const v16bf*)(bs + ((kc * 4 + wid) * 32 + lane) * 16);
            acc = __builtin_amdgcn_wmma_f32_16x16x32_bf16(false, a.v, false, b,
                                                          (short)0, acc, false, false);
        }
        unsigned short* gb = gbuf + (t & 1) * (B_SZ * DH);
        if (lane < 16) {   // lanes 0..15 hold batch rows 0..7; upper lanes = padding
#pragma unroll
            for (int b = 0; b < B_SZ; ++b) {
                size_t idx = ((size_t)b * S_LEN + t) * DH + col;
                float hv = acc[b] + bx[idx];     // h_t = bx_t + h_{t-1} W_A^T
                y[idx] += hv;                    // y = h_t + residual
                gb[b * DH + col] = f2bf_rne(hv); // publish slice for next step
            }
            if (t + 1 < S_LEN)                   // warm L2/L0 for next step's bx
                __builtin_prefetch(bx + (size_t)(t + 1) * DH + col, 0, 3);
        }
        __threadfence();          // make gbuf slice visible device-wide
        __syncthreads();
        if (tid == 0) {           // 16-WG split barrier for step t
            __hip_atomic_fetch_add(&cnt[t], 1u, __ATOMIC_ACQ_REL,
                                   __HIP_MEMORY_SCOPE_AGENT);
            while (__hip_atomic_load(&cnt[t], __ATOMIC_ACQUIRE,
                                     __HIP_MEMORY_SCOPE_AGENT) < (unsigned)SCAN_WGS)
                __builtin_amdgcn_s_sleep(1);
        }
        __syncthreads();
        // Gather full h_t (8x1024 bf16 = 16KB) into LDS for the next step.
        const uint4* gsrc = (const uint4*)gb;
        uint4* hdst = (uint4*)hb;
        for (int i = tid; i < 1024; i += 128) hdst[i] = gsrc[i];
        __syncthreads();
    }
}

// ---------------------------------------------------------------------------
// In-place LayerNorm over the last dim (1024) of y[ROWS][DH].
// ---------------------------------------------------------------------------
__global__ void k_layernorm(float* __restrict__ y,
                            const float* __restrict__ gamma,
                            const float* __restrict__ beta) {
    __shared__ float red[2][8];
    float* p = y + (size_t)blockIdx.x * DH;
    const int tid = threadIdx.x;

    float v[4], s = 0.f, sq = 0.f;
#pragma unroll
    for (int i = 0; i < 4; ++i) {
        v[i] = p[tid + i * 256];
        s += v[i]; sq += v[i] * v[i];
    }
#pragma unroll
    for (int m = 16; m > 0; m >>= 1) {
        s  += __shfl_xor(s,  m, 32);
        sq += __shfl_xor(sq, m, 32);
    }
    const int wid = tid >> 5;
    if ((tid & 31) == 0) { red[0][wid] = s; red[1][wid] = sq; }
    __syncthreads();
    if (wid == 0) {
        float a = (tid < 8) ? red[0][tid] : 0.f;
        float b = (tid < 8) ? red[1][tid] : 0.f;
#pragma unroll
        for (int m = 4; m > 0; m >>= 1) {
            a += __shfl_xor(a, m, 32);
            b += __shfl_xor(b, m, 32);
        }
        if (tid == 0) { red[0][0] = a; red[1][0] = b; }
    }
    __syncthreads();
    float mean = red[0][0] * (1.f / DH);
    float var  = red[1][0] * (1.f / DH) - mean * mean;
    float rstd = __frsqrt_rn(var + 1e-5f);
#pragma unroll
    for (int i = 0; i < 4; ++i) {
        int c = tid + i * 256;
        p[c] = (v[i] - mean) * rstd * gamma[c] + beta[c];
    }
}

// ---------------------------------------------------------------------------
// kernel_launch: inputs in setup_inputs() order:
//   0:x 1:W_B 2:b_B 3:W_A 4:W_res 5:b_res 6:gamma 7:beta   (all fp32)
// ---------------------------------------------------------------------------
extern "C" void kernel_launch(void* const* d_in, const int* in_sizes, int n_in,
                              void* d_out, int out_size, void* d_ws, size_t ws_size,
                              hipStream_t stream) {
    const float* x     = (const float*)d_in[0];
    const float* W_B   = (const float*)d_in[1];
    const float* b_B   = (const float*)d_in[2];
    const float* W_A   = (const float*)d_in[3];
    const float* W_res = (const float*)d_in[4];
    const float* b_res = (const float*)d_in[5];
    const float* gamma = (const float*)d_in[6];
    const float* beta  = (const float*)d_in[7];
    float* y = (float*)d_out;

    // Workspace layout (bytes):
    char* ws = (char*)d_ws;
    float*          bx   = (float*)(ws);                       // 16384*1024*4 = 64MB
    unsigned short* xbf  = (unsigned short*)(ws + 67108864);   // 16384*512*2  = 16MB
    unsigned short* WBsw = (unsigned short*)(ws + 83886080);   // 512*1024*2   =  1MB
    unsigned short* WRsw = (unsigned short*)(ws + 84934656);   // 512*1024*2   =  1MB
    unsigned short* WAsw = (unsigned short*)(ws + 85983232);   // 1024*1024*2  =  2MB
    unsigned short* gbuf = (unsigned short*)(ws + 88080384);   // 2*8*1024*2   = 32KB
    unsigned*       cnt  = (unsigned*)(ws + 88113152);         // 2048*4       =  8KB

    k_f2bf    <<<4096, 256, 0, stream>>>(x, xbf, ROWS * DIN);
    k_swizzleB<<<1024, 256, 0, stream>>>(W_B,   WBsw, DIN);
    k_swizzleB<<<1024, 256, 0, stream>>>(W_res, WRsw, DIN);
    k_swizzleB<<<2048, 256, 0, stream>>>(W_A,   WAsw, DH);
    k_init    <<<32,   256, 0, stream>>>(cnt, gbuf);

    k_gemm_bias<<<16384, 128, 0, stream>>>(xbf, WBsw, b_B,   bx, DIN); // bx
    k_gemm_bias<<<16384, 128, 0, stream>>>(xbf, WRsw, b_res, y,  DIN); // residual -> d_out

    k_scan<<<SCAN_WGS, 128, 163840, stream>>>(WAsw, bx, y, gbuf, cnt);

    k_layernorm<<<ROWS, 256, 0, stream>>>(y, gamma, beta);
}
